// IAF_DDSF_50654844289417
// MI455X (gfx1250) — matrix operand order repaired and checked by reading, not compiled, not run on hardware
//
#include <hip/hip_runtime.h>

// ---------------------------------------------------------------- constants
#define BATCH_N   8192
#define DIM_      64
#define CTX_      32
#define HID_      1920
#define NOUTL_    90
#define NDSP_     49
#define OUTCOLS_  (DIM_ * NOUTL_)   // 5760

#define BM_       128               // block M tile
#define APAD_     40                // padded A-tile row (80B) -> conflict-free b128 reads

#define DELTA_F   1e-6f
#define INV_F     0.5413232766831595f          // log(exp(1-DELTA)-1)
#define LOG1MD_F  (-1.0000005000290891e-06f)   // log(1-DELTA)

typedef __bf16 bf16_t;
typedef __attribute__((ext_vector_type(16))) __bf16 v16bf;
typedef __attribute__((ext_vector_type(8)))  __bf16 v8bf;
typedef __attribute__((ext_vector_type(8)))  float  v8f;
typedef __attribute__((ext_vector_type(4)))  float  v4f;
typedef __attribute__((ext_vector_type(4)))  int    v4i;

#ifndef __has_builtin
#define __has_builtin(x) 0
#endif
#if __has_builtin(__builtin_amdgcn_global_load_async_to_lds_b128)
#define HAVE_ASYNC_LDS 1
typedef __attribute__((address_space(1))) char glob_char;
typedef __attribute__((address_space(3))) char lds_char;
typedef __attribute__((address_space(1))) v4i  gv4i;   // param type: int4 __device__*
typedef __attribute__((address_space(3))) v4i  lv4i;   // param type: int4 __shared__*
#else
#define HAVE_ASYNC_LDS 0
#endif

#if HAVE_ASYNC_LDS
#if __has_builtin(__builtin_amdgcn_s_wait_asynccnt)
#define WAIT_ASYNC(n) __builtin_amdgcn_s_wait_asynccnt(n)
#else
#define WAIT_ASYNC(n) asm volatile("s_wait_asynccnt %0" ::"i"(n) : "memory")
#endif
#endif

// ---------------------------------------------------------------- helpers
__global__ void f32_to_bf16_kernel(const float* __restrict__ in,
                                   bf16_t* __restrict__ out, int n) {
  int i = blockIdx.x * blockDim.x + threadIdx.x;
  if (i < n) out[i] = (bf16_t)in[i];
}

// W[o,k] = g[o] * dir[o,k] * mask[o,k], with optional row L2-normalization of dir
__global__ __launch_bounds__(256)
void prep_w_kernel(const float* __restrict__ dir, const float* __restrict__ g,
                   const float* __restrict__ mask, bf16_t* __restrict__ Wb,
                   int K, int donorm) {
  __shared__ float red[256];
  const int o = blockIdx.x;
  const long base = (long)o * K;
  float inv = 1.0f;
  if (donorm) {
    float s = 0.0f;
    for (int k = threadIdx.x; k < K; k += 256) {
      float d = dir[base + k];
      s = fmaf(d, d, s);
    }
    red[threadIdx.x] = s;
    __syncthreads();
    for (int st = 128; st > 0; st >>= 1) {
      if (threadIdx.x < st) red[threadIdx.x] += red[threadIdx.x + st];
      __syncthreads();
    }
    inv = rsqrtf(red[0]);
  }
  const float gg = g[o] * inv;
  for (int k = threadIdx.x; k < K; k += 256)
    Wb[base + k] = (bf16_t)(gg * dir[base + k] * mask[base + k]);
}

// ---------------------------------------------------------------- fused GEMM + CWN
// C[n,o] = (sum_k A[n,k]*W[o,k] + b[o]) * (ctx[n]·cs_w[o] + cs_b[o])
//                                       + (ctx[n]·ch_w[o] + ch_b[o]);  optional ELU
// Block: 256 threads = 8 waves; tile 128(M) x 128(N); wave = 16-col stripe, 8 M-subtiles.
// A-tile double-buffered in LDS via global_load_async_to_lds (ASYNCcnt pipelined).
template<bool ELU>
__global__ __launch_bounds__(256)
void gemm_cwn_kernel(const bf16_t* __restrict__ A, const bf16_t* __restrict__ W,
                     const float* __restrict__ bias, const float* __restrict__ ctx,
                     const float* __restrict__ cs_w, const float* __restrict__ cs_b,
                     const float* __restrict__ ch_w, const float* __restrict__ ch_b,
                     bf16_t* __restrict__ Hout, int M, int N, int K) {
  __shared__ alignas(16) float  ctxs[BM_][CTX_];       // 16 KB context tile
  __shared__ alignas(16) bf16_t alds[2][BM_][APAD_];   // 2 x 10 KB A tiles

  const int tid  = threadIdx.x;
  const int lane = tid & 31;
  const int wv   = tid >> 5;
  const int m0   = blockIdx.y * BM_;
  const int n0   = blockIdx.x * 128 + wv * 16;

  // stage the 128x32 fp32 context tile once (16 floats / thread)
  {
    const int r = tid >> 1;
    const int c = (tid & 1) * 16;
    const float* src = ctx + (long)(m0 + r) * CTX_ + c;
#pragma unroll
    for (int j = 0; j < 16; j += 4)
      *(v4f*)&ctxs[r][c + j] = *(const v4f*)(src + j);
  }

  // fragment addressing (ISA 7.12.2 layouts, wave32):
  //  A (16x32 bf16):  lane L holds row M=L&15; K = (e/8)*16 + (L/16)*8 + e%8
  //  B (32x16 bf16):  lane L holds col N=L&15; K = (e/8)*16 + (L/16)*8 + e%8
  const int arow = lane & 15;
  const int ak0  = (lane >> 4) << 3;
  const int brow = n0 + (lane & 15);
  const int bk0  = (lane >> 4) << 3;

  v8f acc[8] = {};

  // cooperative A staging: 32B (two b128) per thread per K-step
  const int srow = tid >> 1;
  const int scol = (tid & 1) * 16;
  const bf16_t* sga = A + (long)(m0 + srow) * K + scol;

#if HAVE_ASYNC_LDS
#define STAGE_A(bufi, kk)                                                          \
  do {                                                                             \
    __builtin_amdgcn_global_load_async_to_lds_b128(                                \
        (gv4i*)(glob_char*)(void*)(sga + (kk)),                                    \
        (lv4i*)(lds_char*)(void*)&alds[bufi][srow][scol], 0, 0);                   \
    __builtin_amdgcn_global_load_async_to_lds_b128(                                \
        (gv4i*)(glob_char*)(void*)(sga + (kk) + 8),                                \
        (lv4i*)(lds_char*)(void*)&alds[bufi][srow][scol + 8], 0, 0);               \
  } while (0)
#else
#define STAGE_A(bufi, kk)                                                          \
  do {                                                                             \
    *(v8bf*)&alds[bufi][srow][scol]     = *(const v8bf*)(sga + (kk));              \
    *(v8bf*)&alds[bufi][srow][scol + 8] = *(const v8bf*)(sga + (kk) + 8);          \
  } while (0)
#endif

  int buf = 0;
  STAGE_A(0, 0);
  for (int k = 0; k < K; k += 32) {
    if (k + 32 < K) {      // prefetch next A tile into the other buffer
      STAGE_A(buf ^ 1, k + 32);
#if HAVE_ASYNC_LDS
      WAIT_ASYNC(2);       // oldest tile (current buf) landed in LDS
#endif
    } else {
#if HAVE_ASYNC_LDS
      WAIT_ASYNC(0);
#endif
    }
    __syncthreads();       // all waves' staging visible

    v16bf bf;
    {
      const bf16_t* p = W + (long)brow * K + k + bk0;
      v8bf lo = *(const v8bf*)p;
      v8bf hi = *(const v8bf*)(p + 16);
#pragma unroll
      for (int i = 0; i < 8; ++i) { bf[i] = lo[i]; bf[i + 8] = hi[i]; }
    }
#pragma unroll
    for (int t = 0; t < 8; ++t) {
      const bf16_t* p = &alds[buf][t * 16 + arow][ak0];
      v8bf lo = *(const v8bf*)p;
      v8bf hi = *(const v8bf*)(p + 16);
      v16bf af;
#pragma unroll
      for (int i = 0; i < 8; ++i) { af[i] = lo[i]; af[i + 8] = hi[i]; }
      acc[t] = __builtin_amdgcn_wmma_f32_16x16x32_bf16(
          false, af, false, bf, (short)0, acc[t], false, false);
    }
    __syncthreads();       // done reading buf before it is overwritten next iter
    buf ^= 1;
  }
#undef STAGE_A

  // ---- fused CWN epilogue -------------------------------------------------
  const int col = brow;                          // this lane's output column
  float cw[CTX_], hw[CTX_];
#pragma unroll
  for (int c = 0; c < CTX_; c += 4) {
    v4f a = *(const v4f*)(cs_w + (long)col * CTX_ + c);
    v4f b = *(const v4f*)(ch_w + (long)col * CTX_ + c);
#pragma unroll
    for (int j = 0; j < 4; ++j) { cw[c + j] = a[j]; hw[c + j] = b[j]; }
  }
  const float bb = bias[col];
  const float sb = cs_b[col];
  const float hb = ch_b[col];
  const int rbase = (lane >> 4) << 3;            // D layout: M = r + 8*(lane/16)
#pragma unroll
  for (int t = 0; t < 8; ++t) {
#pragma unroll
    for (int r = 0; r < 8; ++r) {
      const int rl = t * 16 + rbase + r;
      float scale = sb, shift = hb;
#pragma unroll
      for (int c = 0; c < CTX_; c += 4) {
        v4f cc = *(const v4f*)&ctxs[rl][c];      // broadcast LDS read per half-wave
#pragma unroll
        for (int j = 0; j < 4; ++j) {
          scale = fmaf(cc[j], cw[c + j], scale);
          shift = fmaf(cc[j], hw[c + j], shift);
        }
      }
      float v = acc[t][r] + bb;
      v = fmaf(v, scale, shift);
      if (ELU) v = (v > 0.0f) ? v : (__expf(v) - 1.0f);
      Hout[(long)(m0 + rl) * N + col] = (bf16_t)v;
    }
  }
}

// ---------------------------------------------------------------- conv(90->49) + DSF
__device__ __forceinline__ float softplus_f(float v) {
  if (v > 20.0f) return v;
  if (v < -20.0f) return __expf(v);
  return log1pf(__expf(v));
}

// NOTE: sf_u drops out of the math: softmax/log_softmax over the singleton
// axis are identically 1 / 0, so the DSF reduces to a 16-unit scalar flow.
__global__ __launch_bounds__(64)
void dsf_conv_kernel(const bf16_t* __restrict__ outb, const float* __restrict__ x,
                     const float* __restrict__ logdet_in,
                     const float* __restrict__ conv_w, const float* __restrict__ conv_b,
                     const float* __restrict__ sf_w,
                     float* __restrict__ out_x, float* __restrict__ out_ld) {
  __shared__ float red[DIM_];
  const int n = blockIdx.x;
  const int d = threadIdx.x;

  float o[NOUTL_];
  const bf16_t* orow = outb + (long)n * OUTCOLS_ + d * NOUTL_;
#pragma unroll
  for (int k = 0; k < NOUTL_; ++k) o[k] = (float)orow[k];

  float dsp[NDSP_];
  for (int p = 0; p < NDSP_; ++p) {
    float s = conv_b[p];
    const float* cwp = conv_w + p * NOUTL_;
    for (int k = 0; k < NOUTL_; ++k) s = fmaf(o[k], cwp[k], s);
    dsp[p] = s;
  }

  const float xv = x[n * DIM_ + d];

  float ak[16], pwk[16], psk[16];
  float mw = -1e30f;
#pragma unroll
  for (int k = 0; k < 16; ++k) {
    ak[k]  = softplus_f(dsp[k] + INV_F);
    pwk[k] = sf_w[k] + dsp[32 + k];
    psk[k] = fmaf(ak[k], xv, dsp[16 + k]);
    mw = fmaxf(mw, pwk[k]);
  }
  float ew[16], se = 0.0f;
#pragma unroll
  for (int k = 0; k < 16; ++k) { ew[k] = __expf(pwk[k] - mw); se += ew[k]; }
  const float lse = __logf(se);
  const float rse = 1.0f / se;
  float xpre = 0.0f;
#pragma unroll
  for (int k = 0; k < 16; ++k) {
    const float sg = 1.0f / (1.0f + __expf(-psk[k]));
    xpre = fmaf(ew[k] * rse, sg, xpre);
  }
  const float xc  = fmaf(xpre, 1.0f - DELTA_F, 0.5f * DELTA_F);
  const float lxc = __logf(xc);
  const float l1m = log1pf(-xc);
  out_x[n * DIM_ + d] = lxc - l1m;

  float lj[16], mj = -1e30f;
#pragma unroll
  for (int k = 0; k < 16; ++k) {
    lj[k] = (pwk[k] - mw - lse)      // log_softmax(pre_w)
          - softplus_f(-psk[k])      // log_sigmoid(pre_sigm)
          - softplus_f(psk[k])       // log_sigmoid(-pre_sigm)
          + __logf(ak[k]);
    mj = fmaxf(mj, lj[k]);
  }
  float sj = 0.0f;
#pragma unroll
  for (int k = 0; k < 16; ++k) sj += __expf(lj[k] - mj);
  const float ldnd = mj + __logf(sj) + LOG1MD_F - lxc - l1m;

  red[d] = ldnd;
  __syncthreads();
  if (d == 0) {
    float s = 0.0f;
#pragma unroll
    for (int i = 0; i < DIM_; ++i) s += red[i];
    out_ld[n] = s + logdet_in[n];
  }
}

// ---------------------------------------------------------------- launch
extern "C" void kernel_launch(void* const* d_in, const int* in_sizes, int n_in,
                              void* d_out, int out_size, void* d_ws, size_t ws_size,
                              hipStream_t stream) {
  (void)in_sizes; (void)n_in; (void)out_size; (void)ws_size;

  const float* x      = (const float*)d_in[0];
  const float* logdet = (const float*)d_in[1];
  const float* ctx    = (const float*)d_in[2];
  const float *dir[4], *g[4], *b[4], *cs_w[4], *cs_b[4], *ch_w[4], *ch_b[4];
  for (int l = 0; l < 4; ++l) {
    const int base = 3 + 7 * l;   // made_params[l]: dir,g,b,cs_w,cs_b,ch_w,ch_b
    dir[l]  = (const float*)d_in[base + 0];
    g[l]    = (const float*)d_in[base + 1];
    b[l]    = (const float*)d_in[base + 2];
    cs_w[l] = (const float*)d_in[base + 3];
    cs_b[l] = (const float*)d_in[base + 4];
    ch_w[l] = (const float*)d_in[base + 5];
    ch_b[l] = (const float*)d_in[base + 6];
  }
  const float* conv_w = (const float*)d_in[31];
  const float* conv_b = (const float*)d_in[32];
  // d_in[33] = sf_u, mathematically inert (softmax over singleton axis)
  const float* sf_w   = (const float*)d_in[34];
  const float* mask[4] = { (const float*)d_in[35], (const float*)d_in[36],
                           (const float*)d_in[37], (const float*)d_in[38] };

  // workspace carve-up (bf16 buffers), 256B aligned
  char* ws = (char*)d_ws;
  auto take = [&](size_t bytes) -> char* {
    char* p = ws;
    ws += (bytes + 255) & ~(size_t)255;
    return p;
  };
  bf16_t* xb = (bf16_t*)take((size_t)BATCH_N * DIM_ * 2);
  bf16_t* W0 = (bf16_t*)take((size_t)HID_ * DIM_ * 2);
  bf16_t* W1 = (bf16_t*)take((size_t)HID_ * HID_ * 2);
  bf16_t* W2 = (bf16_t*)take((size_t)HID_ * HID_ * 2);
  bf16_t* W3 = (bf16_t*)take((size_t)OUTCOLS_ * HID_ * 2);
  bf16_t* hA = (bf16_t*)take((size_t)BATCH_N * HID_ * 2);
  bf16_t* hB = (bf16_t*)take((size_t)BATCH_N * HID_ * 2);
  bf16_t* ob = (bf16_t*)take((size_t)BATCH_N * OUTCOLS_ * 2);

  f32_to_bf16_kernel<<<(BATCH_N * DIM_) / 256, 256, 0, stream>>>(x, xb, BATCH_N * DIM_);
  prep_w_kernel<<<HID_,     256, 0, stream>>>(dir[0], g[0], mask[0], W0, DIM_, 0);
  prep_w_kernel<<<HID_,     256, 0, stream>>>(dir[1], g[1], mask[1], W1, HID_, 0);
  prep_w_kernel<<<HID_,     256, 0, stream>>>(dir[2], g[2], mask[2], W2, HID_, 0);
  prep_w_kernel<<<OUTCOLS_, 256, 0, stream>>>(dir[3], g[3], mask[3], W3, HID_, 1);

  dim3 blk(256);
  gemm_cwn_kernel<true ><<<dim3(HID_ / 128,     BATCH_N / BM_), blk, 0, stream>>>(
      xb, W0, b[0], ctx, cs_w[0], cs_b[0], ch_w[0], ch_b[0], hA, BATCH_N, HID_, DIM_);
  gemm_cwn_kernel<true ><<<dim3(HID_ / 128,     BATCH_N / BM_), blk, 0, stream>>>(
      hA, W1, b[1], ctx, cs_w[1], cs_b[1], ch_w[1], ch_b[1], hB, BATCH_N, HID_, HID_);
  gemm_cwn_kernel<true ><<<dim3(HID_ / 128,     BATCH_N / BM_), blk, 0, stream>>>(
      hB, W2, b[2], ctx, cs_w[2], cs_b[2], ch_w[2], ch_b[2], hA, BATCH_N, HID_, HID_);
  gemm_cwn_kernel<false><<<dim3(OUTCOLS_ / 128, BATCH_N / BM_), blk, 0, stream>>>(
      hA, W3, b[3], ctx, cs_w[3], cs_b[3], ch_w[3], ch_b[3], ob, BATCH_N, OUTCOLS_, HID_);

  float* out_x  = (float*)d_out;
  float* out_ld = (float*)d_out + (size_t)BATCH_N * DIM_;
  dsf_conv_kernel<<<BATCH_N, DIM_, 0, stream>>>(ob, x, logdet, conv_w, conv_b, sf_w,
                                                out_x, out_ld);
}
